// RowAttentionWithPairBias_62663572848687
// MI455X (gfx1250) — matrix operand order repaired and checked by reading, not compiled, not run on hardware
//
#include <hip/hip_runtime.h>

#define DEVINL __device__ __forceinline__

typedef __attribute__((ext_vector_type(16))) __bf16          v16bf;
typedef __attribute__((ext_vector_type(8)))  float           v8f;
typedef __attribute__((ext_vector_type(8)))  unsigned short  ushort8;

union BF16x16 {
  unsigned short u[16];
  ushort8        h[2];
  v16bf          v;
};

constexpr int NRr = 512;   // rec tokens
constexpr int NAa = 256;   // lig tokens per copy
constexpr int NTt = 768;   // total keys
constexpr int HH  = 8;     // heads
constexpr int MM  = 8;     // lig copies
constexpr int C1c = 256;   // channel dim

DEVINL unsigned short f2bf(float f) {
  union { float f; unsigned u; } x; x.f = f;
  unsigned u = x.u;
  unsigned r = (u + 0x7FFFu + ((u >> 16) & 1u)) >> 16;   // round-nearest-even
  return (unsigned short)r;
}

DEVINL v8f wmma_bf16(const BF16x16& a, const BF16x16& b, v8f c) {
  return __builtin_amdgcn_wmma_f32_16x16x32_bf16(false, a.v, false, b.v,
                                                 (short)0, c, false, false);
}

// A operand (16 rows x 32 K), source row-major [row][k], rows 16B-aligned.
// lane lo: row=lane, K 0..15; lane hi: row=lane-16, K 16..31.
DEVINL BF16x16 load_a16x32(const unsigned short* base, int rowbase, int stride,
                           int kofs, int lane16, int hv) {
  BF16x16 r;
  const unsigned short* p = base + (size_t)(rowbase + lane16) * stride + kofs + hv * 16;
  r.h[0] = *(const ushort8*)(p);
  r.h[1] = *(const ushort8*)(p + 8);
  return r;
}

// B operand (32 K x 16 cols), source stored [col][k] row-major (i.e. B^T).
// lane lo: col=lane, K 0..15; lane hi: col=lane-16, K 16..31.
DEVINL BF16x16 load_b32x16(const unsigned short* base, int colbase, int stride,
                           int kofs, int lane16, int hv) {
  BF16x16 r;
  const unsigned short* p = base + (size_t)(colbase + lane16) * stride + kofs + hv * 16;
  r.h[0] = *(const ushort8*)(p);
  r.h[1] = *(const ushort8*)(p + 8);
  return r;
}

DEVINL v8f load_c_f32(const float* p) {
  float4 a = *(const float4*)p;
  float4 b = *(const float4*)(p + 4);
  v8f c;
  c[0]=a.x; c[1]=a.y; c[2]=a.z; c[3]=a.w;
  c[4]=b.x; c[5]=b.y; c[6]=b.z; c[7]=b.w;
  return c;
}

// ---------------------------------------------------------------- LayerNorm
__global__ __launch_bounds__(256) void ln_bf16_kernel(const float* __restrict__ x,
                                                      const float* __restrict__ g,
                                                      const float* __restrict__ b,
                                                      unsigned short* __restrict__ out) {
  __shared__ float red[256];
  int row = blockIdx.x, t = threadIdx.x;
  float v = x[(size_t)row * C1c + t];
  red[t] = v; __syncthreads();
  for (int s = 128; s > 0; s >>= 1) { if (t < s) red[t] += red[t + s]; __syncthreads(); }
  float mu = red[0] * (1.f / C1c); __syncthreads();
  float d = v - mu;
  red[t] = d * d; __syncthreads();
  for (int s = 128; s > 0; s >>= 1) { if (t < s) red[t] += red[t + s]; __syncthreads(); }
  float var = red[0] * (1.f / C1c);
  float y = d * rsqrtf(var + 1e-5f) * g[t] + b[t];
  out[(size_t)row * C1c + t] = f2bf(y);
}

// ------------------------------------------------- weight transpose to bf16
__global__ __launch_bounds__(256) void transpose_bf16_kernel(const float* __restrict__ W,
                                                             unsigned short* __restrict__ Wt,
                                                             int K, int N) {
  int idx = blockIdx.x * 256 + threadIdx.x;
  if (idx >= K * N) return;
  int k = idx / N, n = idx % N;
  Wt[(size_t)n * K + k] = f2bf(W[idx]);
}

__global__ __launch_bounds__(256) void f32_to_bf16_kernel(const float* __restrict__ x,
                                                          unsigned short* __restrict__ y, int n) {
  int idx = blockIdx.x * 256 + threadIdx.x;
  if (idx < n) y[idx] = f2bf(x[idx]);
}

// -------------------------------------------------------- WMMA bf16 GEMM
// C[M,N] = A[M,K](bf16,row-major) * Bt[N,K](bf16, = B^T) + bias[N]
// 8 waves/block, one 16x16 tile per wave; grid sized exactly.
__global__ __launch_bounds__(256) void gemm_bf16_kernel(const unsigned short* __restrict__ A,
                                                        const unsigned short* __restrict__ Bt,
                                                        const float* __restrict__ bias,
                                                        float* __restrict__ C,
                                                        int Mdim, int Ndim, int Kdim) {
  int wave = threadIdx.x >> 5, lane = threadIdx.x & 31;
  int lane16 = lane & 15, hv = lane >> 4;
  int tiles_n = Ndim >> 4;
  int tile = blockIdx.x * 8 + wave;
  int mt = tile / tiles_n, nt = tile % tiles_n;
  v8f acc = {0.f,0.f,0.f,0.f,0.f,0.f,0.f,0.f};
  const unsigned short* Arow = A  + (size_t)mt * 16 * Kdim;
  const unsigned short* Bcol = Bt + (size_t)nt * 16 * Kdim;
  for (int k = 0; k < Kdim; k += 32) {
    BF16x16 a = load_a16x32(Arow, 0, Kdim, k, lane16, hv);
    BF16x16 b = load_b32x16(Bcol, 0, Kdim, k, lane16, hv);
    acc = wmma_bf16(a, b, acc);
  }
  float bb = bias ? bias[nt * 16 + lane16] : 0.f;
  int col = nt * 16 + lane16;
#pragma unroll
  for (int v = 0; v < 8; ++v) {
    int row = mt * 16 + v + hv * 8;
    C[(size_t)row * Ndim + col] = acc[v] + bb;
  }
}

// ------------------------------------------- repack QKV into WMMA layouts
// qkv[P][1536]; feature f = c*48 + part*8 + h; parts: lq lk lv rq rk rv.
// q/k -> [h][P][32] bf16 ; v -> transposed [h][32][P] (rec) or [h][m][32][256] (lig)
__global__ __launch_bounds__(256) void repack_qkv_kernel(const float* __restrict__ qkv,
    unsigned short* __restrict__ q_l, unsigned short* __restrict__ k_l, unsigned short* __restrict__ v_l,
    unsigned short* __restrict__ q_r, unsigned short* __restrict__ k_r, unsigned short* __restrict__ v_r,
    int P, int isLig) {
  int idx = blockIdx.x * 256 + threadIdx.x;
  if (idx >= P * 1536) return;
  int pos = idx / 1536, f = idx % 1536;
  int c = f / 48, k = f % 48, part = k >> 3, h = k & 7;
  unsigned short val = f2bf(qkv[idx]);
  size_t qk_idx = ((size_t)h * P + pos) * 32 + c;
  size_t v_idx;
  if (isLig) { int m = pos >> 8, j = pos & 255; v_idx = ((size_t)(h * 8 + m) * 32 + c) * 256 + j; }
  else       { v_idx = ((size_t)h * 32 + c) * P + pos; }
  switch (part) {
    case 0: q_l[qk_idx] = val; break;
    case 1: k_l[qk_idx] = val; break;
    case 2: v_l[v_idx]  = val; break;
    case 3: q_r[qk_idx] = val; break;
    case 4: k_r[qk_idx] = val; break;
    default: v_r[v_idx] = val; break;
  }
}

// ---------------------------------------------------- pair bias projection
// biasQK[h][q][k] = (pair[q,k,:] @ W_quad)[h] / sqrt(32). Memory-bound (302MB).
__global__ __launch_bounds__(256) void pair_bias_kernel(const float* __restrict__ pair,
    const float* __restrict__ Wrr, const float* __restrict__ Wll,
    const float* __restrict__ Wrl, const float* __restrict__ Wlr,
    float* __restrict__ biasQK) {
  __shared__ float Ws[128 * 8];
  int t = threadIdx.x;
  size_t cell = (size_t)blockIdx.x * 256 + t;       // q*768 + k, quadrant uniform per block
  int q = (int)(cell / NTt), k = (int)(cell % NTt);
  const float* W = (q < NRr) ? (k < NRr ? Wrr : Wrl) : (k < NRr ? Wlr : Wll);
  for (int i = t; i < 1024; i += 256) Ws[i] = W[i];
  __syncthreads();
  const float* p = pair + cell * 128;
  float acc[8] = {0,0,0,0,0,0,0,0};
  for (int c2 = 0; c2 < 128; ++c2) {
    float pv = p[c2];
#pragma unroll
    for (int h = 0; h < 8; ++h) acc[h] += pv * Ws[c2 * 8 + h];
  }
  const float fac = 0.17677669529663687f;           // 1/sqrt(32)
#pragma unroll
  for (int h = 0; h < 8; ++h)
    biasQK[((size_t)h * NTt + q) * NTt + k] = acc[h] * fac;
}

// ------------------------------------------------------- fused attention
struct AttnCtx {
  const unsigned short* Krec;   // keys<512   [pos][32]
  const unsigned short* Klig;   // keys>=512  [pos][32] (head+m offset)
  const unsigned short* VtRec;  // V^T [c][pos], stride 512
  const unsigned short* VtLig;  // V^T [c][pos], stride 256
  const float* biasRow;         // per-lane bias row base (h,q fixed)
  BF16x16 BqR, BqL;             // Q operands for rec / lig key ranges
};

DEVINL void attn_one_m(const AttnCtx& cx, int p2start,
                       v8f* accR, v8f* accL, int lane16, int hv) {
  // ---- pass 1: softmax stats over all 768 keys (logits = K*Q^T + bias)
  float rmax = -1e30f, rsum = 0.f;
  for (int kb = 0; kb < NTt; kb += 16) {
    bool krec = kb < NRr;
    BF16x16 Ak = krec ? load_a16x32(cx.Krec, kb, 32, 0, lane16, hv)
                      : load_a16x32(cx.Klig, kb - NRr, 32, 0, lane16, hv);
    v8f c = load_c_f32(cx.biasRow + kb + hv * 8);
    c = wmma_bf16(Ak, krec ? cx.BqR : cx.BqL, c);
    float tmax = c[0];
#pragma unroll
    for (int v = 1; v < 8; ++v) tmax = fmaxf(tmax, c[v]);
    float nmax = fmaxf(rmax, tmax);
    rsum *= __expf(rmax - nmax);
#pragma unroll
    for (int v = 0; v < 8; ++v) rsum += __expf(c[v] - nmax);
    rmax = nmax;
  }
  float omax = __shfl_xor(rmax, 16, 32);
  float osum = __shfl_xor(rsum, 16, 32);
  float fmx = fmaxf(rmax, omax);
  float fsum = rsum * __expf(rmax - fmx) + osum * __expf(omax - fmx);
  float inv = 1.0f / fsum;
  // ---- pass 2: recompute logits, form P (bf16), accumulate V^T * P
  for (int kc = p2start; kc < NTt; kc += 32) {
    float p0[8], p1[8];
#pragma unroll
    for (int t = 0; t < 2; ++t) {
      int kb = kc + t * 16;
      bool krec = kb < NRr;
      BF16x16 Ak = krec ? load_a16x32(cx.Krec, kb, 32, 0, lane16, hv)
                        : load_a16x32(cx.Klig, kb - NRr, 32, 0, lane16, hv);
      v8f c = load_c_f32(cx.biasRow + kb + hv * 8);
      c = wmma_bf16(Ak, krec ? cx.BqR : cx.BqL, c);
      float* pp = t ? p1 : p0;
#pragma unroll
      for (int v = 0; v < 8; ++v) pp[v] = __expf(c[v] - fmx) * inv;
    }
    // P (C-layout) -> B operand of AV WMMA: one xor-16 exchange
    BF16x16 Bp;
#pragma unroll
    for (int v = 0; v < 8; ++v) {
      float x0 = __shfl_xor(p0[v], 16, 32);
      float x1 = __shfl_xor(p1[v], 16, 32);
      Bp.u[v]     = f2bf(hv ? x1 : p0[v]);
      Bp.u[8 + v] = f2bf(hv ? p1[v] : x0);
    }
    bool krec = kc < NRr;
    const unsigned short* Vt = krec ? cx.VtRec : cx.VtLig;
    int vs   = krec ? NRr : NAa;
    int kloc = krec ? kc : kc - NRr;
    v8f* acc = krec ? accR : accL;
#pragma unroll
    for (int cc = 0; cc < 2; ++cc) {
      BF16x16 Av = load_a16x32(Vt, cc * 16, vs, kloc, lane16, hv);
      acc[cc] = wmma_bf16(Av, Bp, acc[cc]);
    }
  }
}

__global__ __launch_bounds__(256) void attn_kernel(
    const unsigned short* __restrict__ rq_l, const unsigned short* __restrict__ rk_l,
    const unsigned short* __restrict__ rvT_l,
    const unsigned short* __restrict__ rq_r, const unsigned short* __restrict__ rk_r,
    const unsigned short* __restrict__ rvT_r,
    const unsigned short* __restrict__ lq_l, const unsigned short* __restrict__ lk_l,
    const unsigned short* __restrict__ lvT_l,
    const unsigned short* __restrict__ lq_r, const unsigned short* __restrict__ lk_r,
    const unsigned short* __restrict__ lvT_r,
    const float* __restrict__ biasQK,
    float* __restrict__ attn_rec, float* __restrict__ attn_lig) {
  int wave = threadIdx.x >> 5, lane = threadIdx.x & 31;
  int lane16 = lane & 15, hv = lane >> 4;
  int h = wave;                                     // wave == head
  v8f zero = {0.f,0.f,0.f,0.f,0.f,0.f,0.f,0.f};
  if (blockIdx.x < NRr / 16) {
    // ---- rec query tile: loop all m internally, accumulate rr + mean_m(rl)
    int qb = blockIdx.x * 16;
    AttnCtx cx;
    cx.Krec  = rk_r + (size_t)h * NRr * 32;
    cx.VtRec = rvT_r + (size_t)h * 32 * NRr;
    cx.BqR = load_b32x16(rq_r + (size_t)h * NRr * 32, qb, 32, 0, lane16, hv);
    cx.BqL = load_b32x16(rq_l + (size_t)h * NRr * 32, qb, 32, 0, lane16, hv);
    cx.biasRow = biasQK + ((size_t)h * NTt + qb + lane16) * NTt;
    v8f accR[2] = {zero, zero};
    v8f accL[2] = {zero, zero};
    for (int m = 0; m < MM; ++m) {
      cx.Klig  = lk_r + ((size_t)h * (MM * NAa) + m * NAa) * 32;
      cx.VtLig = lvT_r + (size_t)(h * MM + m) * 32 * NAa;
      attn_one_m(cx, (m == 0) ? 0 : NRr, accR, accL, lane16, hv);
    }
    int q = qb + lane16;
#pragma unroll
    for (int cc = 0; cc < 2; ++cc)
#pragma unroll
      for (int v = 0; v < 8; ++v) {
        int c = cc * 16 + v + hv * 8;
        attn_rec[(size_t)q * C1c + c * HH + h] = accR[cc][v] + 0.125f * accL[cc][v];
      }
  } else {
    // ---- lig query tile for one m
    int idx = blockIdx.x - NRr / 16;
    int m = idx >> 4;
    int qb = (idx & 15) * 16;
    AttnCtx cx;
    cx.Krec  = rk_l + (size_t)h * NRr * 32;
    cx.VtRec = rvT_l + (size_t)h * 32 * NRr;
    cx.Klig  = lk_l + ((size_t)h * (MM * NAa) + m * NAa) * 32;
    cx.VtLig = lvT_l + (size_t)(h * MM + m) * 32 * NAa;
    cx.BqR = load_b32x16(lq_r + ((size_t)h * (MM * NAa) + m * NAa) * 32, qb, 32, 0, lane16, hv);
    cx.BqL = load_b32x16(lq_l + ((size_t)h * (MM * NAa) + m * NAa) * 32, qb, 32, 0, lane16, hv);
    cx.biasRow = biasQK + ((size_t)h * NTt + NRr + qb + lane16) * NTt;
    v8f acc[2] = {zero, zero};
    attn_one_m(cx, 0, acc, acc, lane16, hv);
    int q = qb + lane16;
#pragma unroll
    for (int cc = 0; cc < 2; ++cc)
#pragma unroll
      for (int v = 0; v < 8; ++v) {
        int c = cc * 16 + v + hv * 8;
        attn_lig[((size_t)(m * NAa + q)) * C1c + c * HH + h] = acc[cc][v];
      }
  }
}

// ------------------------------------------------------------------ launch
extern "C" void kernel_launch(void* const* d_in, const int* in_sizes, int n_in,
                              void* d_out, int out_size, void* d_ws, size_t ws_size,
                              hipStream_t stream) {
  (void)in_sizes; (void)n_in; (void)out_size; (void)ws_size;
  const float* rec_profile = (const float*)d_in[0];
  const float* lig_profile = (const float*)d_in[1];
  const float* pair        = (const float*)d_in[2];
  const float* rec_g       = (const float*)d_in[3];
  const float* rec_b       = (const float*)d_in[4];
  const float* lig_g       = (const float*)d_in[5];
  const float* lig_b       = (const float*)d_in[6];
  const float* W_rec_qkv   = (const float*)d_in[7];
  const float* W_lig_qkv   = (const float*)d_in[8];
  const float* W_rr        = (const float*)d_in[9];
  const float* W_ll        = (const float*)d_in[10];
  const float* W_rl        = (const float*)d_in[11];
  const float* W_lr        = (const float*)d_in[12];
  const float* W_rec_fin   = (const float*)d_in[13];
  const float* b_rec_fin   = (const float*)d_in[14];
  const float* W_lig_fin   = (const float*)d_in[15];
  const float* b_lig_fin   = (const float*)d_in[16];
  float* out_rec = (float*)d_out;
  float* out_lig = out_rec + 512 * 256;

  size_t off = 0;
  char* ws = (char*)d_ws;
  auto carve = [&](size_t bytes) -> void* {
    void* p = ws + off;
    off += (bytes + 255) & ~(size_t)255;
    return p;
  };
  unsigned short* rec_ln = (unsigned short*)carve(512  * 256 * 2);
  unsigned short* lig_ln = (unsigned short*)carve(2048 * 256 * 2);
  unsigned short* WrT    = (unsigned short*)carve(1536 * 256 * 2);
  unsigned short* WlT    = (unsigned short*)carve(1536 * 256 * 2);
  unsigned short* WfrT   = (unsigned short*)carve(256  * 256 * 2);
  unsigned short* WflT   = (unsigned short*)carve(256  * 256 * 2);
  float* rqkv = (float*)carve((size_t)512  * 1536 * 4);
  float* lqkv = (float*)carve((size_t)2048 * 1536 * 4);
  unsigned short* rq_l  = (unsigned short*)carve((size_t)8 * 512 * 32 * 2);
  unsigned short* rk_l  = (unsigned short*)carve((size_t)8 * 512 * 32 * 2);
  unsigned short* rvT_l = (unsigned short*)carve((size_t)8 * 512 * 32 * 2);
  unsigned short* rq_r  = (unsigned short*)carve((size_t)8 * 512 * 32 * 2);
  unsigned short* rk_r  = (unsigned short*)carve((size_t)8 * 512 * 32 * 2);
  unsigned short* rvT_r = (unsigned short*)carve((size_t)8 * 512 * 32 * 2);
  unsigned short* lq_l  = (unsigned short*)carve((size_t)8 * 2048 * 32 * 2);
  unsigned short* lk_l  = (unsigned short*)carve((size_t)8 * 2048 * 32 * 2);
  unsigned short* lvT_l = (unsigned short*)carve((size_t)8 * 2048 * 32 * 2);
  unsigned short* lq_r  = (unsigned short*)carve((size_t)8 * 2048 * 32 * 2);
  unsigned short* lk_r  = (unsigned short*)carve((size_t)8 * 2048 * 32 * 2);
  unsigned short* lvT_r = (unsigned short*)carve((size_t)8 * 2048 * 32 * 2);
  float* biasQK     = (float*)carve((size_t)8 * 768 * 768 * 4);
  float* attn_rec_f = (float*)carve((size_t)512  * 256 * 4);
  float* attn_lig_f = (float*)carve((size_t)2048 * 256 * 4);
  unsigned short* attn_rec_h = (unsigned short*)carve((size_t)512  * 256 * 2);
  unsigned short* attn_lig_h = (unsigned short*)carve((size_t)2048 * 256 * 2);

  // 1. LayerNorm -> bf16
  ln_bf16_kernel<<<512,  256, 0, stream>>>(rec_profile, rec_g, rec_b, rec_ln);
  ln_bf16_kernel<<<2048, 256, 0, stream>>>(lig_profile, lig_g, lig_b, lig_ln);
  // 2. weight transposes -> bf16
  transpose_bf16_kernel<<<(256 * 1536) / 256, 256, 0, stream>>>(W_rec_qkv, WrT, 256, 1536);
  transpose_bf16_kernel<<<(256 * 1536) / 256, 256, 0, stream>>>(W_lig_qkv, WlT, 256, 1536);
  transpose_bf16_kernel<<<(256 * 256)  / 256, 256, 0, stream>>>(W_rec_fin, WfrT, 256, 256);
  transpose_bf16_kernel<<<(256 * 256)  / 256, 256, 0, stream>>>(W_lig_fin, WflT, 256, 256);
  // 3. QKV GEMMs (WMMA)
  gemm_bf16_kernel<<<384,  256, 0, stream>>>(rec_ln, WrT, nullptr, rqkv, 512,  1536, 256);
  gemm_bf16_kernel<<<1536, 256, 0, stream>>>(lig_ln, WlT, nullptr, lqkv, 2048, 1536, 256);
  // 4. repack to WMMA-friendly per-head layouts
  repack_qkv_kernel<<<(512  * 1536) / 256, 256, 0, stream>>>(rqkv, rq_l, rk_l, rvT_l, rq_r, rk_r, rvT_r, 512, 0);
  repack_qkv_kernel<<<(2048 * 1536) / 256, 256, 0, stream>>>(lqkv, lq_l, lk_l, lvT_l, lq_r, lk_r, lvT_r, 2048, 1);
  // 5. pair bias (HBM-bound, 302MB)
  pair_bias_kernel<<<(768 * 768) / 256, 256, 0, stream>>>(pair, W_rr, W_ll, W_rl, W_lr, biasQK);
  // 6. fused attention (WMMA QK^T + WMMA PV, flash-style two-pass softmax)
  attn_kernel<<<160, 256, 0, stream>>>(rq_l, rk_l, rvT_l, rq_r, rk_r, rvT_r,
                                       lq_l, lk_l, lvT_l, lq_r, lk_r, lvT_r,
                                       biasQK, attn_rec_f, attn_lig_f);
  // 7. final projections (WMMA)
  f32_to_bf16_kernel<<<(512  * 256) / 256, 256, 0, stream>>>(attn_rec_f, attn_rec_h, 512 * 256);
  f32_to_bf16_kernel<<<(2048 * 256) / 256, 256, 0, stream>>>(attn_lig_f, attn_lig_h, 2048 * 256);
  gemm_bf16_kernel<<<64,  256, 0, stream>>>(attn_rec_h, WfrT, b_rec_fin, out_rec, 512,  256, 256);
  gemm_bf16_kernel<<<256, 256, 0, stream>>>(attn_lig_h, WflT, b_lig_fin, out_lig, 2048, 256, 256);
}
